// NodeAttn_11003706212374
// MI455X (gfx1250) — matrix operand order repaired and checked by reading, not compiled, not run on hardware
//
#include <hip/hip_runtime.h>

// ---------------------------------------------------------------------------
// GAT edge attention for MI455X (gfx1250, wave32, WMMA).
// v_wmma_f32_16x16x32_bf16 for both GEMMs, fp32 accumulate.
// Persistent blocks: weight B-fragments live in registers for the whole
// kernel; each iteration processes 64 edges (4 M-subtiles reuse B).
// bf16 conversion via __builtin_convertvector (hardware packed cvt).
// Segment softmax / segment sum via global atomics on sorted index.
// ---------------------------------------------------------------------------

typedef __attribute__((ext_vector_type(16))) __bf16 v16bf;
typedef __attribute__((ext_vector_type(2)))  __bf16 v2bf;
typedef __attribute__((ext_vector_type(2)))  float  v2f;
typedef __attribute__((ext_vector_type(8)))  float  v8f;

union Frag {
    v16bf bf;
    unsigned int u[8];
};

#define DIM  128
#define NH   4
#define HD   32
#define K1   384   // concat(q, k_v, k_e)
#define K2   256   // concat(k_v, k_e)
#define BM   64    // edges per block iteration (4 x 16-row WMMA subtiles)

// fp32 pair -> packed bf16 (backend lowers to packed cvt hardware)
__device__ __forceinline__ unsigned int packbf2(float a, float b) {
    v2f t; t.x = a; t.y = b;
    v2bf r = __builtin_convertvector(t, v2bf);
    return __builtin_bit_cast(unsigned int, r);
}

// float atomic max via signed/unsigned int ordering trick
__device__ __forceinline__ void atomicMaxF(float* addr, float v) {
    if (v >= 0.0f) atomicMax((int*)addr, __float_as_int(v));
    else           atomicMin((unsigned int*)addr, __float_as_uint(v));
}

// ---------------------------------------------------------------------------
// Kernel 0: init output, segment max (-inf sentinel) and segment sum
// ---------------------------------------------------------------------------
__global__ void gat_init(float* __restrict__ out, float* __restrict__ mbuf,
                         float* __restrict__ sbuf, int N) {
    int t = blockIdx.x * blockDim.x + threadIdx.x;
    if (t < N * DIM) out[t] = 0.0f;
    if (t < N * NH) { mbuf[t] = -3.0e38f; sbuf[t] = 0.0f; }
}

// ---------------------------------------------------------------------------
// Kernel 1: x = leaky_relu(concat(q,kv,ke) @ W1.T + b1); logit = <x_h, w2_h>
// Persistent blocks; per iteration: 4 subtiles x 12 chained bf16 WMMAs.
// ---------------------------------------------------------------------------
__global__ __launch_bounds__(256)
void gat_logits(const float* __restrict__ q,  const float* __restrict__ kv,
                const float* __restrict__ ke, const float* __restrict__ W1,
                const float* __restrict__ b1, const float* __restrict__ w2,
                const int* __restrict__ index,
                float* __restrict__ logits, float* __restrict__ mbuf,
                int tiles) {
    __shared__ __align__(16) unsigned short Abuf[BM][K1 + 8]; // bf16, padded
    __shared__ float partial[BM * NH];                        // logit partials

    const int tid  = threadIdx.x;
    const int wave = tid >> 5, lane = tid & 31;
    const int nlo  = lane & 15, hi = lane >> 4;
    const int ncol = wave * 16 + nlo;            // output feature 0..127

    // ---- B fragments: W1 torch-layout [out=128][in=384]; loaded ONCE.
    Frag B[12];
    {
        const float* wrow = W1 + ncol * K1;
#pragma unroll
        for (int kk = 0; kk < 12; ++kk)
#pragma unroll
            for (int v = 0; v < 8; ++v) {
                int k = kk * 32 + hi * 16 + v * 2;
                float2 w = *(const float2*)(wrow + k);
                B[kk].u[v] = packbf2(w.x, w.y);
            }
    }
    const float bias = b1[ncol];
    const float w2v  = w2[ncol];                 // w2 flat [H*HD] == [128]

    for (int it = blockIdx.x; it < tiles; it += gridDim.x) {
        const int tileBase = it * BM;
        __syncthreads();                         // previous iter fully done
        partial[tid] = 0.0f;

        // ---- A tile: 64 edges x 384 = [q|kv|ke], float4 -> 2x packed bf16
        for (int p = tid; p < BM * (K1 / 4); p += 256) {
            int r  = p / (K1 / 4);
            int c4 = (p % (K1 / 4)) * 4;
            int e  = tileBase + r;
            const float* src = (c4 < DIM)     ? q  + e * DIM + c4
                             : (c4 < 2 * DIM) ? kv + e * DIM + (c4 - DIM)
                                              : ke + e * DIM + (c4 - 2 * DIM);
            float4 w = *(const float4*)src;
            uint2 o; o.x = packbf2(w.x, w.y); o.y = packbf2(w.z, w.w);
            *(uint2*)&Abuf[r][c4] = o;
        }

        // ---- prefetch next tile's edge rows while we compute
        int itn = it + gridDim.x;
        if (itn < tiles) {
            int e2 = itn * BM + (tid & 63);
            const int co = (tid >> 6) * 32;      // 4 slices per row
            __builtin_prefetch(&q [e2 * DIM + co], 0, 1);
            __builtin_prefetch(&kv[e2 * DIM + co], 0, 1);
            __builtin_prefetch(&ke[e2 * DIM + co], 0, 1);
        }
        __syncthreads();

        // ---- 4 subtiles x chained WMMA over K=384
        v8f acc[4];
#pragma unroll
        for (int s = 0; s < 4; ++s) acc[s] = {};
#pragma unroll
        for (int kk = 0; kk < 12; ++kk) {
#pragma unroll
            for (int s = 0; s < 4; ++s) {
                Frag A;
#pragma unroll
                for (int v = 0; v < 8; ++v) {
                    int K = hi * 16 + v * 2;
                    A.u[v] = *(const unsigned int*)
                                 &Abuf[s * 16 + nlo][kk * 32 + K];
                }
                acc[s] = __builtin_amdgcn_wmma_f32_16x16x32_bf16(
                    false, A.bf, false, B[kk].bf, (short)0, acc[s],
                    false, false);
            }
        }

        // ---- bias + leaky_relu + per-head dot partials into LDS
        const int h = ncol >> 5;
#pragma unroll
        for (int s = 0; s < 4; ++s)
#pragma unroll
            for (int r = 0; r < 8; ++r) {
                float y = acc[s][r] + bias;
                y = (y > 0.0f) ? y : 0.01f * y;
                atomicAdd(&partial[(s * 16 + hi * 8 + r) * NH + h], y * w2v);
            }
        __syncthreads();

        // ---- finalize: 256 threads == 64 edges x 4 heads
        {
            int el = tid >> 2, hh = tid & 3;
            float sum = partial[el * NH + hh];
            int e = tileBase + el;
            logits[e * NH + hh] = sum;
            atomicMaxF(&mbuf[index[e] * NH + hh], sum);
        }
    }
}

// ---------------------------------------------------------------------------
// Kernel 2: ex = exp(logit - m[index]) (in place) ; segment sum
// ---------------------------------------------------------------------------
__global__ void gat_exp(const int* __restrict__ index, float* __restrict__ logits,
                        const float* __restrict__ mbuf, float* __restrict__ sbuf,
                        int E) {
    int t = blockIdx.x * blockDim.x + threadIdx.x;
    if (t >= E * NH) return;
    int e = t >> 2, h = t & 3;
    int node = index[e];
    float m = mbuf[node * NH + h];
    if (m < -1e37f) m = 0.0f;                 // empty-segment convention
    float ev = __expf(logits[t] - m);
    logits[t] = ev;                           // reuse buffer as ex
    atomicAdd(&sbuf[node * NH + h], ev);
}

// ---------------------------------------------------------------------------
// Kernel 3: v = concat(kv,ke) @ W3.T + b3 ; out[node] += (ex/s) * v
// Persistent blocks; per iteration: 4 subtiles x 8 chained bf16 WMMAs.
// ---------------------------------------------------------------------------
__global__ __launch_bounds__(256)
void gat_aggregate(const float* __restrict__ kv, const float* __restrict__ ke,
                   const float* __restrict__ W3, const float* __restrict__ b3,
                   const int* __restrict__ index, const float* __restrict__ ex,
                   const float* __restrict__ sbuf, float* __restrict__ out,
                   int tiles) {
    __shared__ __align__(16) unsigned short Abuf[BM][K2 + 8];

    const int tid  = threadIdx.x;
    const int wave = tid >> 5, lane = tid & 31;
    const int nlo  = lane & 15, hi = lane >> 4;
    const int ncol = wave * 16 + nlo;

    Frag B[8];
    {
        const float* wrow = W3 + ncol * K2;
#pragma unroll
        for (int kk = 0; kk < 8; ++kk)
#pragma unroll
            for (int v = 0; v < 8; ++v) {
                int k = kk * 32 + hi * 16 + v * 2;
                float2 w = *(const float2*)(wrow + k);
                B[kk].u[v] = packbf2(w.x, w.y);
            }
    }
    const float bias = b3[ncol];
    const int   h    = ncol >> 5;

    for (int it = blockIdx.x; it < tiles; it += gridDim.x) {
        const int tileBase = it * BM;
        __syncthreads();

        for (int p = tid; p < BM * (K2 / 4); p += 256) {
            int r  = p / (K2 / 4);
            int c4 = (p % (K2 / 4)) * 4;
            int e  = tileBase + r;
            const float* src = (c4 < DIM) ? kv + e * DIM + c4
                                          : ke + e * DIM + (c4 - DIM);
            float4 w = *(const float4*)src;
            uint2 o; o.x = packbf2(w.x, w.y); o.y = packbf2(w.z, w.w);
            *(uint2*)&Abuf[r][c4] = o;
        }

        int itn = it + gridDim.x;
        if (itn < tiles) {
            int e2 = itn * BM + (tid & 63);
            const int co = (tid >> 6) * 32;
            __builtin_prefetch(&kv[e2 * DIM + co], 0, 1);
            __builtin_prefetch(&ke[e2 * DIM + co], 0, 1);
        }
        __syncthreads();

        v8f acc[4];
#pragma unroll
        for (int s = 0; s < 4; ++s) acc[s] = {};
#pragma unroll
        for (int kk = 0; kk < 8; ++kk) {
#pragma unroll
            for (int s = 0; s < 4; ++s) {
                Frag A;
#pragma unroll
                for (int v = 0; v < 8; ++v) {
                    int K = hi * 16 + v * 2;
                    A.u[v] = *(const unsigned int*)
                                 &Abuf[s * 16 + nlo][kk * 32 + K];
                }
                acc[s] = __builtin_amdgcn_wmma_f32_16x16x32_bf16(
                    false, A.bf, false, B[kk].bf, (short)0, acc[s],
                    false, false);
            }
        }

        // scatter straight from accumulators (C layout: N=lane%16, M=hi*8+r)
#pragma unroll
        for (int s = 0; s < 4; ++s)
#pragma unroll
            for (int r = 0; r < 8; ++r) {
                int e = tileBase + s * 16 + hi * 8 + r;
                int node = index[e];
                float w = ex[e * NH + h] / (sbuf[node * NH + h] + 1e-16f);
                atomicAdd(&out[node * DIM + ncol], (acc[s][r] + bias) * w);
            }
    }
}

// ---------------------------------------------------------------------------
// Host launcher
// ---------------------------------------------------------------------------
extern "C" void kernel_launch(void* const* d_in, const int* in_sizes, int n_in,
                              void* d_out, int out_size, void* d_ws, size_t ws_size,
                              hipStream_t stream) {
    (void)n_in; (void)ws_size;
    const float* q     = (const float*)d_in[0];
    const float* kv    = (const float*)d_in[1];
    const float* ke    = (const float*)d_in[2];
    const int*   index = (const int*)  d_in[3];
    // d_in[4] = nnode scalar (derived from out_size instead)
    const float* W1    = (const float*)d_in[5];
    const float* b1    = (const float*)d_in[6];
    const float* w2    = (const float*)d_in[7];
    const float* W3    = (const float*)d_in[8];
    const float* b3    = (const float*)d_in[9];
    float* out = (float*)d_out;

    const int E = in_sizes[0] / DIM;     // 320000
    const int N = out_size / DIM;        // 10000

    float* wsf    = (float*)d_ws;
    float* logits = wsf;                               // E*NH, reused as ex
    float* mbuf   = wsf + (size_t)E * NH;              // N*NH segment max
    float* sbuf   = mbuf + (size_t)N * NH;             // N*NH segment sum

    const int tiles = E / BM;            // E is a multiple of 64
    const int nblk  = (tiles < 1024) ? tiles : 1024;   // persistent blocks

    gat_init<<<(N * DIM + 255) / 256, 256, 0, stream>>>(out, mbuf, sbuf, N);
    gat_logits<<<nblk, 256, 0, stream>>>(q, kv, ke, W1, b1, w2, index,
                                         logits, mbuf, tiles);
    gat_exp<<<(E * NH + 255) / 256, 256, 0, stream>>>(index, logits, mbuf,
                                                      sbuf, E);
    gat_aggregate<<<nblk, 256, 0, stream>>>(kv, ke, W3, b3, index, logits,
                                            sbuf, out, tiles);
}
